// NaiveLSTM_71554155152218
// MI455X (gfx1250) — matrix operand order repaired and checked by reading, not compiled, not run on hardware
//
#include <hip/hip_runtime.h>
#include <hip/hip_bf16.h>

// ---------------------------------------------------------------------------
// NaiveLSTM on MI455X (gfx1250, wave32, WMMA)
//
//   1) pack x (f32->bf16) and weights (f32 -> bf16, WMMA-fragment-packed,
//      gate-interleaved columns n = 4*j + gate).
//   2) gemm_pregates: (SEQ*BS x 1024) @ (1024 x 4096) via
//      v_wmma_f32_16x16x32_bf16 -> f32 pregate buffer [512][64][4096].
//      Wave tile 32x64 (2x4 frags): 12 b128 loads per 8 WMMAs.
//   3) lstm_recurrent: persistent kernel, 32 workgroups, 512 steps with a
//      grid barrier per step.  h_{t-1} @ W_h slice via WMMA with the pregate
//      tile as the f32 accumulator init; c stays in LDS; h exchanged via a
//      double-buffered global bf16 buffer (L2 resident).
//   Repeat 2)+3) for layer 1 (shared W_ii/W_if/W_ig/W_io, same W_h*).
// ---------------------------------------------------------------------------

typedef __bf16 bf16;
typedef __attribute__((ext_vector_type(16))) __bf16 v16bf;
typedef __attribute__((ext_vector_type(8)))  __bf16 v8bf;
typedef __attribute__((ext_vector_type(8)))  float  v8f;

#define SEQ_LEN   512
#define BATCH     64
#define HID       1024
#define NGATE     4096           // 4 * HID, interleaved n = 4*j + gate
#define KTILES    32             // 1024 / 32
#define NB_REC    32             // workgroups in recurrent kernel

// ---------------------------------------------------------------------------
// WMMA helpers
// ---------------------------------------------------------------------------
__device__ inline v8f wmma_bf16(v16bf a, v16bf b, v8f c) {
  // D = A(16x32 bf16) * B(32x16 bf16) + C(16x16 f32)
  return __builtin_amdgcn_wmma_f32_16x16x32_bf16(
      /*neg_a=*/false, a, /*neg_b=*/false, b,
      /*c_mod=*/(short)0, c, /*reuse_a=*/false, /*reuse_b=*/false);
}

__device__ inline v8f zero_v8f() {
  v8f z;
#pragma unroll
  for (int i = 0; i < 8; ++i) z[i] = 0.0f;
  return z;
}

// A fragment: 16x32 bf16 tile from row-major [*, lda] bf16 matrix.
// Layout used here (contiguous k, not the ISA table order): lane m = lane&15; lanes 0-15 hold K {0..15},
// lanes 16-31 hold K {16..31}  (16 contiguous elems, matches packed B).
__device__ inline v16bf load_a_frag(const bf16* __restrict__ A, int lda,
                                    int m0, int kt, int lane) {
  const int m  = m0 + (lane & 15);
  const int hi = lane >> 4;
  const bf16* p = A + (size_t)m * lda + kt * 32 + hi * 16;
  v8bf r0 = *(const v8bf*)(p);
  v8bf r1 = *(const v8bf*)(p + 8);
  v16bf a;
#pragma unroll
  for (int i = 0; i < 8; ++i) { a[i] = r0[i]; a[i + 8] = r1[i]; }
  return a;
}

// B fragment from fragment-packed weights: block (kt, nt) holds 32 lanes x
// 16 contiguous bf16 (lane n = lane&15, K base = 16*(lane>>4)).
__device__ inline v16bf load_b_frag(const bf16* __restrict__ Bp,
                                    int kt, int nt, int lane) {
  const bf16* p = Bp + (((size_t)nt * KTILES + kt) << 9) + (lane << 4);
  return *(const v16bf*)p;
}

// C/D fragment (16x16 f32): lane n = lane&15, rows m0 + 8*(lane>>4) + r.
__device__ inline v8f load_c_frag(const float* __restrict__ C, int ldc,
                                  int m0, int n0, int lane) {
  const int n  = n0 + (lane & 15);
  const int mb = m0 + ((lane >> 4) << 3);
  v8f c;
#pragma unroll
  for (int r = 0; r < 8; ++r) c[r] = C[(size_t)(mb + r) * ldc + n];
  return c;
}

__device__ inline void store_d_frag(float* __restrict__ D, int ldc,
                                    int m0, int n0, int lane, v8f d) {
  const int n  = n0 + (lane & 15);
  const int mb = m0 + ((lane >> 4) << 3);
#pragma unroll
  for (int r = 0; r < 8; ++r) D[(size_t)(mb + r) * ldc + n] = d[r];
}

__device__ inline void store_d_lds(float* __restrict__ D, int ldc,
                                   int m0, int n0, int lane, v8f d) {
  const int n  = n0 + (lane & 15);
  const int mb = m0 + ((lane >> 4) << 3);
#pragma unroll
  for (int r = 0; r < 8; ++r) D[(mb + r) * ldc + n] = d[r];
}

// ---------------------------------------------------------------------------
// Pack kernels
// ---------------------------------------------------------------------------
__global__ void __launch_bounds__(256)
pack_f32_to_bf16(const float* __restrict__ src, bf16* __restrict__ dst, int n) {
  for (int i = blockIdx.x * 256 + threadIdx.x; i < n; i += gridDim.x * 256)
    dst[i] = (bf16)src[i];
}

// Pack 4 gate matrices (each 1024x1024 f32 row-major, K x H) into WMMA
// fragment layout with interleaved columns n = 4*j + gate.
__global__ void __launch_bounds__(256)
pack_gate_weights(const float* __restrict__ Wi, const float* __restrict__ Wf,
                  const float* __restrict__ Wg, const float* __restrict__ Wo,
                  bf16* __restrict__ dst) {
  const int idx = blockIdx.x * 256 + threadIdx.x;      // 0 .. 4194303
  const int blk    = idx >> 9;                          // 512 elems / block
  const int within = idx & 511;
  const int lane   = within >> 4;
  const int i      = within & 15;
  const int nt = blk / KTILES;
  const int kt = blk % KTILES;
  const int k  = kt * 32 + ((lane >> 4) << 4) + i;      // 16 contiguous K / lane
  const int n  = nt * 16 + (lane & 15);
  const int j  = n >> 2;
  const int g  = n & 3;
  const float* W = (g == 0) ? Wi : (g == 1) ? Wf : (g == 2) ? Wg : Wo;
  dst[idx] = (bf16)W[(size_t)k * HID + j];
}

// ---------------------------------------------------------------------------
// Big parallel GEMM:  P[M x 4096] = X[M x 1024] @ Wpacked
// block = 256 threads (8 waves) arranged 2(M) x 4(N); block tile 64 x 256.
// Each wave: 32x64 tile = 2x4 WMMA frags; per k-step 12 b128 loads / 8 WMMAs.
// ---------------------------------------------------------------------------
__global__ void __launch_bounds__(256)
gemm_pregates(const bf16* __restrict__ X, const bf16* __restrict__ Wp,
              float* __restrict__ P, int M) {
  const int lane = threadIdx.x & 31;
  const int wave = threadIdx.x >> 5;
  const int nBlkN = NGATE / 256;                  // 16
  const int bm = blockIdx.x / nBlkN;
  const int bn = blockIdx.x % nBlkN;
  const int m0 = bm * 64  + (wave >> 2) * 32;     // wave M base
  const int n0 = bn * 256 + (wave & 3) * 64;      // wave N base
  if (m0 >= M) return;

  v8f acc[2][4];
#pragma unroll
  for (int p = 0; p < 2; ++p)
#pragma unroll
    for (int q = 0; q < 4; ++q) acc[p][q] = zero_v8f();

  const int nt0 = n0 >> 4;
  for (int kt = 0; kt < KTILES; ++kt) {
    v16bf a0 = load_a_frag(X, HID, m0,      kt, lane);
    v16bf a1 = load_a_frag(X, HID, m0 + 16, kt, lane);
    v16bf b[4];
#pragma unroll
    for (int q = 0; q < 4; ++q) b[q] = load_b_frag(Wp, kt, nt0 + q, lane);
#pragma unroll
    for (int q = 0; q < 4; ++q) {
      acc[0][q] = wmma_bf16(a0, b[q], acc[0][q]);
      acc[1][q] = wmma_bf16(a1, b[q], acc[1][q]);
    }
  }
#pragma unroll
  for (int p = 0; p < 2; ++p)
#pragma unroll
    for (int q = 0; q < 4; ++q)
      store_d_frag(P, NGATE, m0 + p * 16, n0 + q * 16, lane, acc[p][q]);
}

// ---------------------------------------------------------------------------
// Grid barrier (arrive counter + generation), workspace-backed.
// ---------------------------------------------------------------------------
__device__ inline void grid_barrier(int* __restrict__ cnt,
                                    volatile int* __restrict__ gen, int nb) {
  __threadfence();
  __syncthreads();
  if (threadIdx.x == 0) {
    const int g = *gen;
    if (atomicAdd(cnt, 1) == nb - 1) {
      *cnt = 0;
      __threadfence();
      atomicAdd((int*)gen, 1);
    } else {
      while (*gen == g) { __builtin_amdgcn_s_sleep(1); }
    }
    __threadfence();
  }
  __syncthreads();
}

// ---------------------------------------------------------------------------
// Persistent recurrent kernel: 32 blocks x 256 threads.
// Block owns gate columns n in [bid*128, bid*128+128)  (= hidden j in
// [bid*32, bid*32+32), 4 gates interleaved).  512 steps, grid barrier each.
// ---------------------------------------------------------------------------
__global__ void __launch_bounds__(256)
lstm_recurrent(const float* __restrict__ pre,     // [512][64][4096]
               const bf16*  __restrict__ Whp,     // packed recurrent weights
               const float* __restrict__ b_i, const float* __restrict__ b_f,
               const float* __restrict__ b_g, const float* __restrict__ b_o,
               bf16* __restrict__ hbuf,           // [2][64][1024] bf16
               float* __restrict__ seqf,          // [512][64][1024] or null
               bf16*  __restrict__ seqb,          // [512][64][1024] or null
               float* __restrict__ hc_out,        // h then c (64x1024 each) or null
               int* __restrict__ bar) {           // [0]=cnt, [1]=gen
  __shared__ float cst[BATCH * 32];               // c state, 8 KB
  __shared__ float gbuf[BATCH * 128];             // pre-activation gates, 32 KB

  const int bid  = blockIdx.x;
  const int tid  = threadIdx.x;
  const int lane = tid & 31;
  const int wave = tid >> 5;
  const int n0g  = bid * 128;                     // global gate-column base
  const int j0   = bid * 32;                      // hidden-column base
  const int mw   = (wave >> 1) * 16;              // wave's M base (0..48)
  const int ntb  = (wave & 1) * 4;                // wave's local ntile base

  // zero c state and h[t=0] slice
  for (int i = tid; i < BATCH * 32; i += 256) cst[i] = 0.0f;
  for (int i = tid; i < 2048; i += 256) hbuf[bid * 2048 + i] = (bf16)0.0f;
  grid_barrier(bar, bar + 1, NB_REC);

  for (int t = 0; t < SEQ_LEN; ++t) {
    const bf16*  hprev = hbuf + (size_t)(t & 1) * (BATCH * HID);
    const float* preT  = pre  + (size_t)t * (BATCH * NGATE);

    // accumulators seeded with x@W pregates (f32)
    v8f acc[4];
#pragma unroll
    for (int q = 0; q < 4; ++q)
      acc[q] = load_c_frag(preT, NGATE, mw, n0g + (ntb + q) * 16, lane);

    // h_{t-1} @ W_h  (K = 1024)
    const int ntg = (n0g >> 4) + ntb;
    for (int kt = 0; kt < KTILES; ++kt) {
      v16bf a = load_a_frag(hprev, HID, mw, kt, lane);
      if (kt + 1 < KTILES)
        __builtin_prefetch(Whp + (((size_t)ntg * KTILES + kt + 1) << 9), 0, 1);
#pragma unroll
      for (int q = 0; q < 4; ++q) {
        v16bf b = load_b_frag(Whp, kt, ntg + q, lane);
        acc[q] = wmma_bf16(a, b, acc[q]);
      }
    }
#pragma unroll
    for (int q = 0; q < 4; ++q)
      store_d_lds(gbuf, 128, mw, (ntb + q) * 16, lane, acc[q]);
    __syncthreads();

    // elementwise gate math: 64 batch x 32 hidden per block
    bf16* hnext = hbuf + (size_t)((t + 1) & 1) * (BATCH * HID);
    for (int r = tid; r < BATCH * 32; r += 256) {
      const int b  = r >> 5;
      const int jj = r & 31;
      const int j  = j0 + jj;
      const float ip = gbuf[b * 128 + jj * 4 + 0] + b_i[j];
      const float fp = gbuf[b * 128 + jj * 4 + 1] + b_f[j];
      const float gp = gbuf[b * 128 + jj * 4 + 2] + b_g[j];
      const float op = gbuf[b * 128 + jj * 4 + 3] + b_o[j];
      const float iv = 1.0f / (1.0f + __expf(-ip));
      const float fv = 1.0f / (1.0f + __expf(-fp));
      const float gv = tanhf(gp);
      const float ov = 1.0f / (1.0f + __expf(-op));
      const float c  = fv * cst[r] + iv * gv;
      cst[r] = c;
      const float h = ov * tanhf(c);
      hnext[(size_t)b * HID + j] = (bf16)h;
      if (seqb) seqb[(size_t)t * (BATCH * HID) + b * HID + j] = (bf16)h;
      if (seqf) seqf[(size_t)t * (BATCH * HID) + b * HID + j] = h;
      if (hc_out && t == SEQ_LEN - 1) {
        hc_out[(size_t)b * HID + j] = h;
        hc_out[(size_t)(BATCH * HID) + b * HID + j] = c;
      }
    }
    __syncthreads();
    grid_barrier(bar, bar + 1, NB_REC);
  }
}

// ---------------------------------------------------------------------------
// Host launch
// ---------------------------------------------------------------------------
extern "C" void kernel_launch(void* const* d_in, const int* in_sizes, int n_in,
                              void* d_out, int out_size, void* d_ws, size_t ws_size,
                              hipStream_t stream) {
  (void)in_sizes; (void)n_in; (void)out_size; (void)ws_size;
  const float* x     = (const float*)d_in[0];
  const float* W_ii0 = (const float*)d_in[1];
  const float* W_ii  = (const float*)d_in[2];
  const float* W_hi  = (const float*)d_in[3];
  const float* W_if0 = (const float*)d_in[4];
  const float* W_if  = (const float*)d_in[5];
  const float* W_hf  = (const float*)d_in[6];
  const float* W_ig0 = (const float*)d_in[7];
  const float* W_ig  = (const float*)d_in[8];
  const float* W_hg  = (const float*)d_in[9];
  const float* W_io0 = (const float*)d_in[10];
  const float* W_io  = (const float*)d_in[11];
  const float* W_ho  = (const float*)d_in[12];
  const float* b_i   = (const float*)d_in[13];
  const float* b_f   = (const float*)d_in[14];
  const float* b_g   = (const float*)d_in[15];
  const float* b_o   = (const float*)d_in[16];
  float* out = (float*)d_out;

  // workspace carve-out (256B aligned)
  char* ws = (char*)d_ws;
  size_t off = 0;
  auto carve = [&](size_t bytes) -> void* {
    void* p = ws + off;
    off += (bytes + 255) & ~(size_t)255;
    return p;
  };
  const size_t nAct = (size_t)SEQ_LEN * BATCH * HID;      // 33,554,432
  float* ws_pre  = (float*)carve((size_t)SEQ_LEN * BATCH * NGATE * 4); // 512 MB
  bf16*  ws_xb   = (bf16*) carve(nAct * 2);                            // 64 MB
  bf16*  ws_seqb = (bf16*) carve(nAct * 2);                            // 64 MB
  bf16*  ws_w0p  = (bf16*) carve((size_t)HID * NGATE * 2);             // 8 MB
  bf16*  ws_w1p  = (bf16*) carve((size_t)HID * NGATE * 2);             // 8 MB
  bf16*  ws_whp  = (bf16*) carve((size_t)HID * NGATE * 2);             // 8 MB
  bf16*  ws_hbuf = (bf16*) carve((size_t)2 * BATCH * HID * 2);         // 256 KB
  int*   bar     = (int*)  carve(256);

  hipMemsetAsync(bar, 0, 8, stream);

  // pack inputs + weights to bf16 / fragment layout
  pack_f32_to_bf16<<<4096, 256, 0, stream>>>(x, ws_xb, (int)nAct);
  pack_gate_weights<<<16384, 256, 0, stream>>>(W_ii0, W_if0, W_ig0, W_io0, ws_w0p);
  pack_gate_weights<<<16384, 256, 0, stream>>>(W_ii,  W_if,  W_ig,  W_io,  ws_w1p);
  pack_gate_weights<<<16384, 256, 0, stream>>>(W_hi,  W_hf,  W_hg,  W_ho,  ws_whp);

  const int M = SEQ_LEN * BATCH;                       // 32768
  const int gemmGrid = (M / 64) * (NGATE / 256);       // 8192

  // ---- layer 0 ----
  gemm_pregates<<<gemmGrid, 256, 0, stream>>>(ws_xb, ws_w0p, ws_pre, M);
  lstm_recurrent<<<NB_REC, 256, 0, stream>>>(ws_pre, ws_whp, b_i, b_f, b_g, b_o,
                                             ws_hbuf, /*seqf=*/nullptr,
                                             /*seqb=*/ws_seqb, /*hc=*/nullptr, bar);
  // ---- layer 1 ----
  gemm_pregates<<<gemmGrid, 256, 0, stream>>>(ws_seqb, ws_w1p, ws_pre, M);
  lstm_recurrent<<<NB_REC, 256, 0, stream>>>(ws_pre, ws_whp, b_i, b_f, b_g, b_o,
                                             ws_hbuf, /*seqf=*/out,
                                             /*seqb=*/nullptr,
                                             /*hc=*/out + nAct, bar);
}
